// Transformer_36146444763553
// MI455X (gfx1250) — compile-verified, matrix operand order checked by reading, not run on hardware
//
// CDNA5 / gfx1250 transformer forward (compile-only target).
// Only layer l=1 of enc/dec matters (reference loops overwrite x_out/x_final).
// All GEMM-shaped math -> v_wmma_f32_16x16x32_bf16 (bf16 in, f32 accum);
// LN / softmax / rel-bias / bias+GELU epilogues stay fp32.
// GEMM tile staging: Tensor Data Mover (tensor_load_to_lds, TENSORcnt) when the
// builtin exists, with hardware row padding matched to the 80B LDS row stride;
// manual global_load_b128 + global_prefetch fallback otherwise.
#include <hip/hip_runtime.h>
#include <hip/hip_bf16.h>
#include <cstdint>
#include <cstddef>

#define DV    512
#define HH    8
#define HSV   64
#define TV    256
#define BV    8
#define VV    32000
#define DFFV  5120
#define BTV   (BV * TV)      // 2048 rows
#define EPSV  1e-5f
#define SCALEV 8.0f          // sqrt(HS)

typedef __bf16 bf16_t;
typedef __attribute__((ext_vector_type(16))) __bf16 bf16x16;
typedef __attribute__((ext_vector_type(8)))  float  v8f;

union Frag { uint4 u[2]; bf16x16 v; };

__device__ __forceinline__ v8f wmma_bf16(bf16x16 a, bf16x16 b, v8f c) {
  // (neg_a, A, neg_b, B, c_mod, C, reuse_a, reuse_b)
  return __builtin_amdgcn_wmma_f32_16x16x32_bf16(false, a, false, b, (short)0, c,
                                                 false, false);
}

// ------------------------------------------------------------------ TDM plumbing
#if defined(__has_builtin)
#if __has_builtin(__builtin_amdgcn_tensor_load_to_lds)
#define USE_TDM 1
#endif
#endif

#ifdef USE_TDM
typedef unsigned int u32x4 __attribute__((ext_vector_type(4)));
typedef int          i32x4 __attribute__((ext_vector_type(4)));
typedef int          i32x8 __attribute__((ext_vector_type(8)));

// D# for one [128 rows x 64B] bf16 tile, row stride Kd elements in memory,
// LDS destination padded +16B per 64B row (-> 80B rows = 40 halves).
__device__ __forceinline__ void tdm_tile_load(const bf16_t* gsrc, unsigned lds_byte,
                                              int Kd) {
  unsigned long long ga = (unsigned long long)(uintptr_t)gsrc;
  u32x4 g0;
  g0.x = 1u;                                   // count=1, user descriptor
  g0.y = lds_byte;                             // lds_addr
  g0.z = (unsigned)(ga & 0xFFFFFFFFu);         // global_addr[31:0]
  g0.w = (unsigned)((ga >> 32) & 0x01FFFFFFu)  // global_addr[56:32]
         | (2u << 30);                         // type = 2 ("image")
  i32x8 g1;
  g1.s0 = (1 << 16)                            // data_size = 2 bytes
        | (1 << 20)                            // pad_enable
        | (3 << 22)                            // pad_interval: 16 DWORDs (64B)
        | (3 << 25);                           // pad_amount: 4 DWORDs (16B)
  g1.s1 = (32 & 0xFFFF) << 16;                 // tensor_dim0[15:0] = 32 elems
  g1.s2 = (128 & 0xFFFF) << 16;                // tensor_dim1[15:0] = 128 rows
  g1.s3 = 32 << 16;                            // tile_dim0 = 32 elems (64B)
  g1.s4 = 128;                                 // tile_dim1 = 128 rows
  g1.s5 = Kd;                                  // tensor_dim0_stride[31:0]
  g1.s6 = 0;
  g1.s7 = 0;
  i32x4 gz = {0, 0, 0, 0};                     // 2D tensor: groups 2/3 unused
#if __clang_major__ >= 23
  i32x8 gz8 = {0, 0, 0, 0, 0, 0, 0, 0};
  __builtin_amdgcn_tensor_load_to_lds(g0, g1, gz, gz, gz8, 0);
#else
  __builtin_amdgcn_tensor_load_to_lds(g0, g1, gz, gz, 0);
#endif
}
#endif // USE_TDM

// ---------------------------------------------------------------- embed
__global__ void k_embed(const int* __restrict__ idx, const float* __restrict__ tok,
                        const float* __restrict__ pos, float* __restrict__ xf,
                        bf16_t* __restrict__ xb) {
  size_t i = (size_t)blockIdx.x * blockDim.x + threadIdx.x;
  if (i >= (size_t)BTV * DV) return;
  int d  = (int)(i % DV);
  int bt = (int)(i / DV);
  int t  = bt % TV;
  float v = tok[(size_t)idx[bt] * DV + d] + pos[(size_t)t * DV + d];
  xf[i] = v;
  xb[i] = (bf16_t)v;
}

// ------------------------------------------------- weight pack: out[b][n][k] = in[b][k][n], f32->bf16
__global__ void k_packw(const float* __restrict__ in, bf16_t* __restrict__ out,
                        int Kd, int Nd, long total) {
  long i = (long)blockIdx.x * blockDim.x + threadIdx.x;
  if (i >= total) return;
  long kn = (long)Kd * Nd;
  long b  = i / kn;
  long r  = i - b * kn;
  long n  = r / Kd;
  long k  = r - n * Kd;
  out[i] = (bf16_t)in[b * kn + k * Nd + n];
}

// ---------------------------------------------------------------- add (+ optional bf16 copy)
__global__ void k_add(const float* __restrict__ a, const float* __restrict__ b,
                      float* __restrict__ cf, bf16_t* __restrict__ cb, int n) {
  int i = blockIdx.x * blockDim.x + threadIdx.x;
  if (i >= n) return;
  float v = a[i] + b[i];
  cf[i] = v;
  if (cb) cb[i] = (bf16_t)v;
}

// ---------------------------------------------------------------- layernorm over D=512
__global__ __launch_bounds__(256) void k_ln(const float* __restrict__ x,
                                            const float* __restrict__ g,
                                            const float* __restrict__ bb,
                                            float* __restrict__ yf,
                                            bf16_t* __restrict__ yb) {
  __shared__ float red[256];
  int row = blockIdx.x, tid = threadIdx.x;
  const float* xr = x + (size_t)row * DV;
  float x0 = xr[tid], x1 = xr[tid + 256];
  red[tid] = x0 + x1;
  __syncthreads();
  for (int o = 128; o > 0; o >>= 1) { if (tid < o) red[tid] += red[tid + o]; __syncthreads(); }
  float mean = red[0] * (1.0f / DV);
  __syncthreads();
  float d0 = x0 - mean, d1 = x1 - mean;
  red[tid] = d0 * d0 + d1 * d1;
  __syncthreads();
  for (int o = 128; o > 0; o >>= 1) { if (tid < o) red[tid] += red[tid + o]; __syncthreads(); }
  float rstd = rsqrtf(red[0] * (1.0f / DV) + EPSV);
  size_t base = (size_t)row * DV;
  float v0 = d0 * rstd * g[tid] + bb[tid];
  float v1 = d1 * rstd * g[tid + 256] + bb[tid + 256];
  if (yf) { yf[base + tid] = v0; yf[base + tid + 256] = v1; }
  if (yb) { yb[base + tid] = (bf16_t)v0; yb[base + tid + 256] = (bf16_t)v1; }
}

// ---------------------------------------------------------------- row softmax over T=256
__global__ __launch_bounds__(256) void k_softmax(const float* __restrict__ S,
                                                 bf16_t* __restrict__ P, int causal) {
  __shared__ float red[256];
  int row = blockIdx.x, tid = threadIdx.x;
  int t = row % TV;
  float x = S[(size_t)row * TV + tid];
  if (causal && tid > t) x = -INFINITY;
  red[tid] = x; __syncthreads();
  for (int o = 128; o > 0; o >>= 1) { if (tid < o) red[tid] = fmaxf(red[tid], red[tid + o]); __syncthreads(); }
  float mx = red[0]; __syncthreads();
  float e = expf(x - mx);
  red[tid] = e; __syncthreads();
  for (int o = 128; o > 0; o >>= 1) { if (tid < o) red[tid] += red[tid + o]; __syncthreads(); }
  float inv = 1.0f / red[0];
  P[(size_t)row * TV + tid] = (bf16_t)(e * inv);
}

// ----------------------------------- rel bias: S[b,h,t,v] += dot(Q[b,h,t,:], rel[h,t,v,:])
// one block per (t,h); rel rows read exactly once, reused across all B via LDS-cached q.
__global__ __launch_bounds__(256) void k_rel(const float* __restrict__ Qf,
                                             const float* __restrict__ rel,
                                             float* __restrict__ S) {
  __shared__ float qs[BV][HSV];
  int t = blockIdx.x, h = blockIdx.y, tid = threadIdx.x;
  for (int i = tid; i < BV * HSV; i += 256) {
    int b = i >> 6, c = i & 63;
    qs[b][c] = Qf[((size_t)(b * TV + t)) * DV + h * HSV + c];
  }
  __syncthreads();
  int v = tid;
  const float* rp = rel + (((size_t)h * TV + t) * TV + v) * HSV;
  float acc[BV] = {};
  #pragma unroll 8
  for (int c = 0; c < HSV; ++c) {
    float r = rp[c];
    #pragma unroll
    for (int b = 0; b < BV; ++b) acc[b] += qs[b][c] * r;
  }
  #pragma unroll
  for (int b = 0; b < BV; ++b)
    S[(((size_t)(b * HH + h)) * TV + t) * TV + v] += acc[b];
}

// ----------------------------------- V transpose: Vt[b,h,c,v] = V[b,v,h,c] (f32->bf16)
__global__ void k_vt(const float* __restrict__ Vf, bf16_t* __restrict__ Vt) {
  size_t i = (size_t)blockIdx.x * blockDim.x + threadIdx.x;
  if (i >= (size_t)BV * HH * HSV * TV) return;
  int v = (int)(i % TV);
  int c = (int)((i / TV) % HSV);
  int h = (int)((i / ((size_t)TV * HSV)) % HH);
  int b = (int)(i / ((size_t)TV * HSV * HH));
  Vt[i] = (bf16_t)Vf[((size_t)(b * TV + v)) * DV + h * HSV + c];
}

// ----------------------------------- scores: S[b,h,t,v] = SCALE * sum_c Q*K  (WMMA)
// grid (T/16, H, B), 128 threads = 4 waves; each wave does 4 v-tiles.
__global__ __launch_bounds__(128) void k_scores(const bf16_t* __restrict__ Q,
                                                const bf16_t* __restrict__ Kb,
                                                float* __restrict__ S) {
  int tt = blockIdx.x, h = blockIdx.y, b = blockIdx.z;
  int wid = threadIdx.x >> 5, lane = threadIdx.x & 31;
  int rsel = lane & 15, kg = lane >> 4;
  int t = tt * 16 + rsel;
  size_t qbase = ((size_t)(b * TV + t)) * DV + h * HSV + kg * 8;
  Frag a0, a1;
  a0.u[0] = *(const uint4*)(Q + qbase);
  a0.u[1] = *(const uint4*)(Q + qbase + 16);
  a1.u[0] = *(const uint4*)(Q + qbase + 32);
  a1.u[1] = *(const uint4*)(Q + qbase + 48);
  size_t sbase = (((size_t)(b * HH + h)) * TV + tt * 16) * TV;
  for (int i = 0; i < 4; ++i) {
    int vt = wid * 4 + i;
    int v  = vt * 16 + rsel;
    size_t kbase = ((size_t)(b * TV + v)) * DV + h * HSV + kg * 8;
    Frag b0, b1;
    b0.u[0] = *(const uint4*)(Kb + kbase);
    b0.u[1] = *(const uint4*)(Kb + kbase + 16);
    b1.u[0] = *(const uint4*)(Kb + kbase + 32);
    b1.u[1] = *(const uint4*)(Kb + kbase + 48);
    v8f acc = {};
    acc = wmma_bf16(a0.v, b0.v, acc);
    acc = wmma_bf16(a1.v, b1.v, acc);
    #pragma unroll
    for (int r = 0; r < 8; ++r) {
      int m = r + kg * 8;
      S[sbase + (size_t)m * TV + vt * 16 + rsel] = acc[r] * SCALEV;
    }
  }
}

// ----------------------------------- O[b,t,h*64+c] = sum_v P[b,h,t,v] * Vt[b,h,c,v] (WMMA)
// grid (T/16, H, B), 128 threads = 4 waves; wave = c-tile.
__global__ __launch_bounds__(128) void k_av(const bf16_t* __restrict__ P,
                                            const bf16_t* __restrict__ Vt,
                                            bf16_t* __restrict__ O) {
  int tt = blockIdx.x, h = blockIdx.y, b = blockIdx.z;
  int wid = threadIdx.x >> 5, lane = threadIdx.x & 31;
  int rsel = lane & 15, kg = lane >> 4;
  int t = tt * 16 + rsel;
  int c = wid * 16 + rsel;
  size_t pbase = (((size_t)(b * HH + h)) * TV + t) * TV + kg * 8;
  size_t vbase = (((size_t)(b * HH + h)) * HSV + c) * TV + kg * 8;
  v8f acc = {};
  for (int k0 = 0; k0 < TV; k0 += 32) {
    Frag a, bb;
    a.u[0]  = *(const uint4*)(P + pbase + k0);
    a.u[1]  = *(const uint4*)(P + pbase + k0 + 16);
    bb.u[0] = *(const uint4*)(Vt + vbase + k0);
    bb.u[1] = *(const uint4*)(Vt + vbase + k0 + 16);
    acc = wmma_bf16(a.v, bb.v, acc);
  }
  #pragma unroll
  for (int r = 0; r < 8; ++r) {
    int m = tt * 16 + r + kg * 8;
    O[((size_t)(b * TV + m)) * DV + h * HSV + wid * 16 + rsel] = (bf16_t)acc[r];
  }
}

// ----------------------------------- generic GEMM: C[M,N] = A[M,K](bf16) * Bt[N,K]^T + bias
// 128x128x32 tiles, 8 waves as 2(M)x4(N), each wave 4x2 WMMA tiles.
// Tile staging via Tensor Data Mover (row-padded DMA) when available.
__global__ __launch_bounds__(256) void k_gemm(const bf16_t* __restrict__ A,
                                              const bf16_t* __restrict__ Bt,
                                              const float* __restrict__ bias,
                                              float* __restrict__ Cf,
                                              bf16_t* __restrict__ Cb,
                                              int M, int N, int Kd, int act) {
  __shared__ __align__(16) bf16_t As[128 * 40];  // stride 40 halves = 80B (64B row + 16B pad)
  __shared__ __align__(16) bf16_t Bs[128 * 40];
  int tid = threadIdx.x;
  int wid = tid >> 5, lane = tid & 31;
  int rsel = lane & 15, kg = lane >> 4;
  int m0 = blockIdx.y * 128, n0 = blockIdx.x * 128;
  int waveM = wid >> 2, waveN = wid & 3;
  v8f acc[4][2];
  #pragma unroll
  for (int i = 0; i < 4; ++i)
    #pragma unroll
    for (int j = 0; j < 2; ++j) { v8f z = {}; acc[i][j] = z; }

  for (int k0 = 0; k0 < Kd; k0 += 32) {
#ifdef USE_TDM
    // One wave drives the TDM; hardware pads each 64B row by 16B to LDS stride 80B.
    if (wid == 0) {
      tdm_tile_load(A  + (size_t)m0 * Kd + k0, (unsigned)(uintptr_t)(void*)As, Kd);
      tdm_tile_load(Bt + (size_t)n0 * Kd + k0, (unsigned)(uintptr_t)(void*)Bs, Kd);
      __builtin_amdgcn_s_wait_tensorcnt(0);
    }
#else
    #pragma unroll
    for (int p = 0; p < 2; ++p) {
      int lin = p * 256 + tid;
      int row = lin >> 2, cg = (lin & 3) * 8;
      const bf16_t* ga = A  + (size_t)(m0 + row) * Kd + k0 + cg;
      const bf16_t* gb = Bt + (size_t)(n0 + row) * Kd + k0 + cg;
      *(uint4*)(As + row * 40 + cg) = *(const uint4*)ga;
      *(uint4*)(Bs + row * 40 + cg) = *(const uint4*)gb;
      if (k0 + 32 < Kd) {                 // pull next K-slab toward L2 (global_prefetch)
        __builtin_prefetch(ga + 32, 0, 0);
        __builtin_prefetch(gb + 32, 0, 0);
      }
    }
#endif
    __syncthreads();
    Frag af[4], bfr[2];
    #pragma unroll
    for (int ms = 0; ms < 4; ++ms) {
      int r = waveM * 64 + ms * 16 + rsel;
      af[ms].u[0] = *(const uint4*)(As + r * 40 + kg * 8);
      af[ms].u[1] = *(const uint4*)(As + r * 40 + 16 + kg * 8);
    }
    #pragma unroll
    for (int ns = 0; ns < 2; ++ns) {
      int r = waveN * 32 + ns * 16 + rsel;
      bfr[ns].u[0] = *(const uint4*)(Bs + r * 40 + kg * 8);
      bfr[ns].u[1] = *(const uint4*)(Bs + r * 40 + 16 + kg * 8);
    }
    #pragma unroll
    for (int ms = 0; ms < 4; ++ms)
      #pragma unroll
      for (int ns = 0; ns < 2; ++ns)
        acc[ms][ns] = wmma_bf16(af[ms].v, bfr[ns].v, acc[ms][ns]);
    __syncthreads();
  }

  #pragma unroll
  for (int ns = 0; ns < 2; ++ns) {
    int n = n0 + waveN * 32 + ns * 16 + rsel;
    float bv = bias ? bias[n] : 0.0f;
    #pragma unroll
    for (int ms = 0; ms < 4; ++ms) {
      int mb = m0 + waveM * 64 + ms * 16 + kg * 8;
      #pragma unroll
      for (int r = 0; r < 8; ++r) {
        float vv = acc[ms][ns][r] + bv;
        if (act) vv = 0.5f * vv * (1.0f + erff(vv * 0.70710678118f));  // exact GELU
        size_t o = (size_t)(mb + r) * N + n;
        if (Cf) Cf[o] = vv;
        if (Cb) Cb[o] = (bf16_t)vv;
      }
    }
  }
}

// ================================================================ host side
namespace {

struct Scratch {
  float *xf, *xout, *t0, *t1, *t2, *t3, *Qf, *Vf, *S;
  bf16_t *xb, *bb0, *bb1, *Qb, *Kb2, *Vt, *P, *Ob, *h1b;
  bf16_t *wqkv_t, *wo_t, *w1t, *w2t, *wft;
};

inline void run_gemm(hipStream_t st, const bf16_t* A, const bf16_t* Bt, const float* bias,
                     float* Cf, bf16_t* Cb, int M, int N, int K, int act) {
  dim3 g(N / 128, M / 128);
  k_gemm<<<g, 256, 0, st>>>(A, Bt, bias, Cf, Cb, M, N, K, act);
}

// one fused-head MHA; writes attention output (fp32) into out_f
inline void run_mha(hipStream_t st, const Scratch& w, const bf16_t* xb,
                    const float* Wq, const float* bq, const float* Wk, const float* bk,
                    const float* Wv, const float* rel, const float* Wo, const float* bo,
                    int causal, float* out_f) {
  const long tw = (long)HH * HSV * DV;           // per-head [D,HS] -> [HS,D] transposes
  // Q
  k_packw<<<(unsigned)((tw + 255) / 256), 256, 0, st>>>(Wq, w.wqkv_t, DV, HSV, tw);
  run_gemm(st, xb, w.wqkv_t, bq, w.Qf, w.Qb, BTV, DV, DV, 0);
  // K
  k_packw<<<(unsigned)((tw + 255) / 256), 256, 0, st>>>(Wk, w.wqkv_t, DV, HSV, tw);
  run_gemm(st, xb, w.wqkv_t, bk, nullptr, w.Kb2, BTV, DV, DV, 0);
  // V (no bias)
  k_packw<<<(unsigned)((tw + 255) / 256), 256, 0, st>>>(Wv, w.wqkv_t, DV, HSV, tw);
  run_gemm(st, xb, w.wqkv_t, nullptr, w.Vf, nullptr, BTV, DV, DV, 0);
  const long tv = (long)BV * HH * HSV * TV;
  k_vt<<<(unsigned)((tv + 255) / 256), 256, 0, st>>>(w.Vf, w.Vt);
  // scores + rel + softmax + AV
  dim3 gs(TV / 16, HH, BV);
  k_scores<<<gs, 128, 0, st>>>(w.Qb, w.Kb2, w.S);
  dim3 gr(TV, HH);
  k_rel<<<gr, 256, 0, st>>>(w.Qf, rel, w.S);
  k_softmax<<<BV * HH * TV, 256, 0, st>>>(w.S, w.P, causal);
  k_av<<<gs, 128, 0, st>>>(w.P, w.Vt, w.Ob);
  // output projection
  const long two = (long)DV * DV;
  k_packw<<<(unsigned)((two + 255) / 256), 256, 0, st>>>(Wo, w.wo_t, DV, DV, two);
  run_gemm(st, w.Ob, w.wo_t, bo, out_f, nullptr, BTV, DV, DV, 0);
}

} // namespace

extern "C" void kernel_launch(void* const* d_in, const int* in_sizes, int n_in,
                              void* d_out, int out_size, void* d_ws, size_t ws_size,
                              hipStream_t stream) {
  (void)in_sizes; (void)n_in; (void)out_size; (void)ws_size;
  const int*   idx = (const int*)d_in[0];
  const float* tok = (const float*)d_in[1];
  const float* pos = (const float*)d_in[2];
  auto F = [&](int i) { return (const float*)d_in[i]; };
  // block layout: Wq bq Wk bk Wv rel Wo bo W1 b1 W2 b2 ln1g ln1b ln2g ln2b
  const int EB = 3, DB = 19;
  const float* lnf_g = F(35); const float* lnf_b = F(36);
  const float* Wf = F(37);   const float* bf_ = F(38);

  // --- layer-1 parameter views (only the last layer's output survives in the ref)
  const size_t oW  = (size_t)HH * DV * HSV;      // Wq/Wk/Wv per-layer stride
  const size_t oRel = (size_t)HH * TV * TV * HSV;
  auto blk = [&](int base, int j) { return F(base + j); };
  struct Blk {
    const float *Wq, *bq, *Wk, *bk, *Wv, *rel, *Wo, *bo, *W1, *b1, *W2, *b2,
                *ln1g, *ln1b, *ln2g, *ln2b;
  };
  auto mk = [&](int base) {
    Blk p;
    p.Wq  = blk(base, 0)  + oW;          p.bq  = blk(base, 1)  + HH * HSV;
    p.Wk  = blk(base, 2)  + oW;          p.bk  = blk(base, 3)  + HH * HSV;
    p.Wv  = blk(base, 4)  + oW;          p.rel = blk(base, 5)  + oRel;
    p.Wo  = blk(base, 6)  + (size_t)DV * DV;  p.bo = blk(base, 7) + DV;
    p.W1  = blk(base, 8)  + (size_t)DV * DFFV; p.b1 = blk(base, 9) + DFFV;
    p.W2  = blk(base, 10) + (size_t)DFFV * DV; p.b2 = blk(base, 11) + DV;
    p.ln1g = blk(base, 12) + DV; p.ln1b = blk(base, 13) + DV;
    p.ln2g = blk(base, 14) + DV; p.ln2b = blk(base, 15) + DV;
    return p;
  };
  Blk enc = mk(EB), dec = mk(DB);

  // --- workspace bump allocator
  char* wsp = (char*)d_ws;
  size_t off = 0;
  auto alloc = [&](size_t bytes) -> void* {
    void* p = wsp + off;
    off += (bytes + 255) & ~(size_t)255;
    return p;
  };
  Scratch w;
  const size_t nBD = (size_t)BTV * DV;
  w.xf   = (float*)alloc(nBD * 4);  w.xb  = (bf16_t*)alloc(nBD * 2);
  w.xout = (float*)alloc(nBD * 4);
  w.t0 = (float*)alloc(nBD * 4); w.t1 = (float*)alloc(nBD * 4);
  w.t2 = (float*)alloc(nBD * 4); w.t3 = (float*)alloc(nBD * 4);
  w.bb0 = (bf16_t*)alloc(nBD * 2); w.bb1 = (bf16_t*)alloc(nBD * 2);
  w.Qf = (float*)alloc(nBD * 4); w.Qb = (bf16_t*)alloc(nBD * 2);
  w.Kb2 = (bf16_t*)alloc(nBD * 2);
  w.Vf = (float*)alloc(nBD * 4); w.Vt = (bf16_t*)alloc(nBD * 2);
  const size_t nS = (size_t)BV * HH * TV * TV;
  w.S = (float*)alloc(nS * 4);  w.P = (bf16_t*)alloc(nS * 2);
  w.Ob = (bf16_t*)alloc(nBD * 2);
  w.h1b = (bf16_t*)alloc((size_t)BTV * DFFV * 2);
  w.wqkv_t = (bf16_t*)alloc((size_t)HH * HSV * DV * 2);
  w.wo_t   = (bf16_t*)alloc((size_t)DV * DV * 2);
  w.w1t = (bf16_t*)alloc((size_t)DV * DFFV * 2);
  w.w2t = (bf16_t*)alloc((size_t)DFFV * DV * 2);
  w.wft = (bf16_t*)alloc((size_t)DV * VV * 2);

  hipStream_t st = stream;
  const int nAdd = BTV * DV;

  // ---------- embed
  k_embed<<<(unsigned)((nBD + 255) / 256), 256, 0, st>>>(idx, tok, pos, w.xf, w.xb);

  // ---------- encoder (l=1)
  run_mha(st, w, w.xb, enc.Wq, enc.bq, enc.Wk, enc.bk, enc.Wv, enc.rel, enc.Wo, enc.bo,
          /*causal=*/0, w.t0);
  k_add<<<nAdd / 256, 256, 0, st>>>(w.xf, w.t0, w.t1, nullptr, nAdd);         // x + attn
  k_ln<<<BTV, 256, 0, st>>>(w.t1, enc.ln1g, enc.ln1b, w.t2, w.bb0);           // src2
  {
    long t1n = (long)DV * DFFV;
    k_packw<<<(unsigned)((t1n + 255) / 256), 256, 0, st>>>(enc.W1, w.w1t, DV, DFFV, t1n);
    run_gemm(st, w.bb0, w.w1t, enc.b1, nullptr, w.h1b, BTV, DFFV, DV, /*gelu=*/1);
    long t2n = (long)DFFV * DV;
    k_packw<<<(unsigned)((t2n + 255) / 256), 256, 0, st>>>(enc.W2, w.w2t, DFFV, DV, t2n);
    run_gemm(st, w.h1b, w.w2t, enc.b2, w.t0, nullptr, BTV, DV, DFFV, 0);
  }
  k_add<<<nAdd / 256, 256, 0, st>>>(w.t2, w.t0, w.t1, nullptr, nAdd);         // src2 + ffn
  k_ln<<<BTV, 256, 0, st>>>(w.t1, enc.ln2g, enc.ln2b, w.xout, nullptr);       // x_out

  // ---------- decoder (l=1)
  run_mha(st, w, w.xb, dec.Wq, dec.bq, dec.Wk, dec.bk, dec.Wv, dec.rel, dec.Wo, dec.bo,
          /*causal=*/1, w.t0);
  k_add<<<nAdd / 256, 256, 0, st>>>(w.xf, w.t0, w.t1, nullptr, nAdd);         // s1
  k_ln<<<BTV, 256, 0, st>>>(w.t1, dec.ln1g, dec.ln1b, w.t2, nullptr);         // ln1(s1)
  k_add<<<nAdd / 256, 256, 0, st>>>(w.t1, w.t2, w.t3, nullptr, nAdd);         // s2
  k_add<<<nAdd / 256, 256, 0, st>>>(w.t3, w.xout, w.t1, w.bb0, nAdd);         // att1 (f32+bf16)
  run_mha(st, w, w.bb0, dec.Wq, dec.bq, dec.Wk, dec.bk, dec.Wv, dec.rel, dec.Wo, dec.bo,
          /*causal=*/0, w.t0);
  k_add<<<nAdd / 256, 256, 0, st>>>(w.t1, w.t0, w.t2, nullptr, nAdd);         // att2
  k_ln<<<BTV, 256, 0, st>>>(w.t2, dec.ln1g, dec.ln1b, w.t0, nullptr);         // ln1(att2)
  k_add<<<nAdd / 256, 256, 0, st>>>(w.t2, w.t0, w.t3, nullptr, nAdd);         // trg
  k_ln<<<BTV, 256, 0, st>>>(w.t3, dec.ln2g, dec.ln2b, nullptr, w.bb0);        // ln2(trg) bf16
  {
    long t1n = (long)DV * DFFV;
    k_packw<<<(unsigned)((t1n + 255) / 256), 256, 0, st>>>(dec.W1, w.w1t, DV, DFFV, t1n);
    run_gemm(st, w.bb0, w.w1t, dec.b1, nullptr, w.h1b, BTV, DFFV, DV, /*gelu=*/1);
    long t2n = (long)DFFV * DV;
    k_packw<<<(unsigned)((t2n + 255) / 256), 256, 0, st>>>(dec.W2, w.w2t, DFFV, DV, t2n);
    run_gemm(st, w.h1b, w.w2t, dec.b2, w.t0, nullptr, BTV, DV, DFFV, 0);
  }
  k_add<<<nAdd / 256, 256, 0, st>>>(w.t3, w.t0, w.t1, nullptr, nAdd);         // src_f
  k_ln<<<BTV, 256, 0, st>>>(w.t1, dec.ln2g, dec.ln2b, w.t2, nullptr);         // decoder out

  // ---------- final LN + vocab projection (dominant GEMM: 2048x32000x512)
  k_ln<<<BTV, 256, 0, st>>>(w.t2, lnf_g, lnf_b, nullptr, w.bb1);
  long tfn = (long)DV * VV;
  k_packw<<<(unsigned)((tfn + 255) / 256), 256, 0, st>>>(Wf, w.wft, DV, VV, tfn);
  run_gemm(st, w.bb1, w.wft, bf_, (float*)d_out, nullptr, BTV, VV, DV, 0);
}
